// CausalLM_RNN_42460046688384
// MI455X (gfx1250) — matrix operand order repaired and checked by reading, not compiled, not run on hardware
//
#include <hip/hip_runtime.h>
#include <hip/hip_bf16.h>

// ---------------------------------------------------------------------------
// Problem constants (from reference): V=32000, E=H=1024, B=4, T=1024, 2 layers
// ---------------------------------------------------------------------------
#define CV   32000
#define CH   1024
#define CB   4
#define CT   1024
#define CM   (CB * CT)     // 4096 rows for the big GEMMs
#define CG   (4 * CH)      // 4096 gate outputs
#define NWG_LSTM 32        // persistent workgroups in recurrence kernel

// LDS layout for the recurrence kernel: Whh slice 128 rows x (1024+8) halves
// (+8 pad => row stride 2064B => ds_load_b128 lanes land on distinct banks)
#define LDS_W    (CH + 8)
#define LDS_WHH_HALVES (128 * LDS_W)
#define LSTM_SMEM_BYTES (LDS_WHH_HALVES * 2 + 4 * CB * 32 * 4)

typedef __attribute__((ext_vector_type(16))) __bf16 v16bf;
typedef __attribute__((ext_vector_type(8)))  float  v8f;

union Frag32B { uint4 q[2]; v16bf v; };

__device__ __forceinline__ unsigned short f32_to_bf16_rne(float f) {
    unsigned u = __float_as_uint(f);
    u += 0x7FFFu + ((u >> 16) & 1u);           // round-to-nearest-even
    return (unsigned short)(u >> 16);
}

// A fragment: 16x32 bf16 tile at (m0, k0) of row-major A (lda elems).
// lane<16: M=lane, K = k0..k0+15; lane>=16: K = k0+16..k0+31.
__device__ __forceinline__ v16bf load_a_frag(const unsigned short* A, int lda,
                                             int m0, int k0, int lane) {
    int m   = lane & 15;
    int hi8 = (lane >> 4) << 4;
    const unsigned short* p = A + (size_t)(m0 + m) * lda + k0 + hi8;
    Frag32B f;
    f.q[0] = *reinterpret_cast<const uint4*>(p);        // K  0..7 (+hi16)
    f.q[1] = *reinterpret_cast<const uint4*>(p + 8);    // K  8..15 (+hi16)
    return f.v;
}

// B fragment: 32x16 tile = rows n0..n0+15 of W (row-major N x K), transposed use.
// lane<16: N=lane, K = k0..k0+15 ; lane>=16: N=lane-16, K = k0+16..k0+31.
__device__ __forceinline__ v16bf load_b_frag(const unsigned short* W, int ldw,
                                             int n0, int k0, int lane) {
    int n    = lane & 15;
    int hi16 = (lane >> 4) << 4;
    const unsigned short* p = W + (size_t)(n0 + n) * ldw + k0 + hi16;
    Frag32B f;
    f.q[0] = *reinterpret_cast<const uint4*>(p);
    f.q[1] = *reinterpret_cast<const uint4*>(p + 8);
    return f.v;
}

// Same B fragment but from the padded LDS-resident slice (local row index).
__device__ __forceinline__ v16bf load_b_frag_lds(const unsigned short* Wl,
                                                 int row0, int k0, int lane) {
    int n    = lane & 15;
    int hi16 = (lane >> 4) << 4;
    const unsigned short* p = Wl + (row0 + n) * LDS_W + k0 + hi16;
    Frag32B f;
    f.q[0] = *reinterpret_cast<const uint4*>(p);
    f.q[1] = *reinterpret_cast<const uint4*>(p + 8);
    return f.v;
}

__device__ __forceinline__ float sigmoid_f(float x) {
    return 1.0f / (1.0f + __expf(-x));
}
__device__ __forceinline__ float tanh_f(float x) {
    return 1.0f - 2.0f / (__expf(2.0f * x) + 1.0f);   // tanh(x)
}

// ---------------------------------------------------------------------------
// f32 -> bf16 conversion (grid-stride)
// ---------------------------------------------------------------------------
__global__ void k_cvt_bf16(const float* __restrict__ s, unsigned short* __restrict__ d, int n) {
    for (int i = blockIdx.x * blockDim.x + threadIdx.x; i < n; i += gridDim.x * blockDim.x)
        d[i] = f32_to_bf16_rne(s[i]);
}

// ---------------------------------------------------------------------------
// Embedding gather -> bf16 rows (one block per token)
// ---------------------------------------------------------------------------
__global__ void k_gather(const int* __restrict__ x, const float* __restrict__ embed,
                         unsigned short* __restrict__ X) {
    int row = blockIdx.x;                      // 0..CM-1  (= b*T + t)
    int id  = x[row];
    const float* src = embed + (size_t)id * CH;
    unsigned short* dst = X + (size_t)row * CH;
    for (int e = threadIdx.x; e < CH; e += blockDim.x)
        dst[e] = f32_to_bf16_rne(src[e]);
}

// ---------------------------------------------------------------------------
// GEMM: C[M x N] f32 = A[M x 1024] bf16 * W[N x 1024]^T bf16 (+ b1[n] + b2[n])
// block = 256 threads (8 wave32) stacked in M: WG tile 128M x 128N,
// wave tile 16M x 128N (8 accumulators). B frags shared via WGP$/L2.
// ---------------------------------------------------------------------------
__global__ __launch_bounds__(256)
void k_gemm(const unsigned short* __restrict__ A, const unsigned short* __restrict__ W,
            const float* __restrict__ b1, const float* __restrict__ b2,
            float* __restrict__ C, int N) {
    constexpr int K = CH;
    int lane = threadIdx.x & 31;
    int w    = threadIdx.x >> 5;
    int m0   = blockIdx.y * 128 + w * 16;
    int n0   = blockIdx.x * 128;

    v8f acc[8] = {v8f{}, v8f{}, v8f{}, v8f{}, v8f{}, v8f{}, v8f{}, v8f{}};
    for (int k0 = 0; k0 < K; k0 += 32) {
        v16bf a = load_a_frag(A, K, m0, k0, lane);
#pragma unroll
        for (int j = 0; j < 8; ++j) {
            v16bf b = load_b_frag(W, K, n0 + j * 16, k0, lane);
            acc[j] = __builtin_amdgcn_wmma_f32_16x16x32_bf16(
                         false, a, false, b, (short)0, acc[j], false, false);
        }
    }
    int hi = (lane >> 4) << 3;
    int nc = lane & 15;
#pragma unroll
    for (int j = 0; j < 8; ++j) {
        int n = n0 + j * 16 + nc;
        float bias = b1[n] + (b2 ? b2[n] : 0.0f);
#pragma unroll
        for (int v = 0; v < 8; ++v) {
            int m = m0 + v + hi;
            C[(size_t)m * N + n] = acc[j][v] + bias;
        }
    }
}

// ---------------------------------------------------------------------------
// Persistent LSTM recurrence for one layer.
//   PRE   : [B*T, 4H] f32  (x@Wih^T + bih + bhh), row r = b*T + t
//   Whh   : [4H, H] bf16
//   hstate: [16, H] bf16   (rows 0..3 = batch h_t, rows 4..15 stay zero)
//   Hout  : [B*T, H] bf16
// 32 workgroups x 256 threads. WG owns units u in [wg*32, wg*32+32).
// Its Whh slice (rows {g*H + wg*32 .. +31, g=0..3} = 128 x 1024 bf16, 256KB)
// is staged into LDS once; every step reads B-frags from LDS (ds_load_b128)
// and only the 8KB h-state from L2. One grid barrier per timestep.
// ---------------------------------------------------------------------------
__global__ __launch_bounds__(256)
void k_lstm(const float* __restrict__ PRE, const unsigned short* __restrict__ Whh,
            unsigned short* __restrict__ hstate, unsigned short* __restrict__ Hout,
            unsigned* __restrict__ bar) {
    extern __shared__ __align__(16) unsigned short smem[];
    unsigned short* wlds = smem;                                   // [128][LDS_W]
    float* zs = (float*)(smem + LDS_WHH_HALVES);                   // [4][CB][32]

    const int tid  = threadIdx.x;
    const int lane = tid & 31;
    const int w    = tid >> 5;
    const int wg   = blockIdx.x;
    const int gate = w >> 1, half = w & 1;
    const int lrow0 = gate * 32 + half * 16;   // local row base in LDS slice

    // ---- stage this WG's Whh slice into LDS (16384 uint4 transfers) ----
    for (int idx = tid; idx < 128 * (CH / 8); idx += 256) {
        int lr = idx >> 7;                      // local row 0..127
        int q  = idx & 127;                     // uint4 index within row
        int g  = lr >> 5, i = lr & 31;
        const uint4* src = reinterpret_cast<const uint4*>(
            Whh + (size_t)(g * CH + wg * 32 + i) * CH + q * 8);
        *reinterpret_cast<uint4*>(wlds + lr * LDS_W + q * 8) = *src;
    }
    __syncthreads();

    const int b  = tid & 3;                    // valid for tid < 128
    const int uu = tid >> 2;
    float c_reg  = 0.0f;
    unsigned nbar = 0;

#pragma unroll 1
    for (int t = 0; t < CT; ++t) {
        // ---- z-slice = h_{t-1} @ Whh^T for this wave's 16 gate rows ----
        v8f acc = v8f{};
#pragma unroll 4
        for (int kt = 0; kt < CH; kt += 32) {
            v16bf a  = load_a_frag(hstate, CH, 0, kt, lane);
            v16bf bb = load_b_frag_lds(wlds, lrow0, kt, lane);
            acc = __builtin_amdgcn_wmma_f32_16x16x32_bf16(
                      false, a, false, bb, (short)0, acc, false, false);
        }
        if (lane < 16) {                       // rows 0..3 = real batch
            int col = half * 16 + lane;        // 0..31 within this wg
#pragma unroll
            for (int v = 0; v < CB; ++v)
                zs[(gate * CB + v) * 32 + col] = acc[v];
        }
        __syncthreads();

        // ---- gate fusion + cell update for this wg's 32 units x 4 batch ----
        if (tid < CB * 32) {
            int u = wg * 32 + uu;
            size_t r = (size_t)b * CT + t;
            const float* pr = PRE + r * CG;
            float iz = zs[(0 * CB + b) * 32 + uu] + pr[u];
            float fz = zs[(1 * CB + b) * 32 + uu] + pr[CH + u];
            float gz = zs[(2 * CB + b) * 32 + uu] + pr[2 * CH + u];
            float oz = zs[(3 * CB + b) * 32 + uu] + pr[3 * CH + u];
            float ig = sigmoid_f(iz);
            float fg = sigmoid_f(fz);
            float gg = tanh_f(gz);
            float og = sigmoid_f(oz);
            c_reg = fg * c_reg + ig * gg;
            float h = og * tanh_f(c_reg);
            unsigned short hb = f32_to_bf16_rne(h);
            hstate[(size_t)b * CH + u] = hb;
            Hout[r * CH + u] = hb;
        }

        // ---- grid-wide barrier: h_t visible to every WGP before t+1 ----
        __builtin_amdgcn_fence(__ATOMIC_RELEASE, "agent");
        __syncthreads();
        if (tid == 0) {
            ++nbar;
            atomicAdd(bar, 1u);
            while (__hip_atomic_load(bar, __ATOMIC_RELAXED, __HIP_MEMORY_SCOPE_AGENT)
                   < nbar * NWG_LSTM)
                __builtin_amdgcn_s_sleep(1);
        }
        __syncthreads();
        __builtin_amdgcn_fence(__ATOMIC_ACQUIRE, "agent");
    }
}

// ---------------------------------------------------------------------------
// Host-side orchestration
// ---------------------------------------------------------------------------
extern "C" void kernel_launch(void* const* d_in, const int* in_sizes, int n_in,
                              void* d_out, int out_size, void* d_ws, size_t ws_size,
                              hipStream_t stream) {
    const int*   x     = (const int*)  d_in[0];   // (B,T)
    const float* embed = (const float*)d_in[1];   // (V,E)
    const float* Wproj = (const float*)d_in[2];   // (V,H)
    const float* bproj = (const float*)d_in[3];   // (V)
    const float* Wih0  = (const float*)d_in[4];   // (4H,E)
    const float* Whh0  = (const float*)d_in[5];   // (4H,H)
    const float* bih0  = (const float*)d_in[6];
    const float* bhh0  = (const float*)d_in[7];
    const float* Wih1  = (const float*)d_in[8];   // (4H,H)
    const float* Whh1  = (const float*)d_in[9];
    const float* bih1  = (const float*)d_in[10];
    const float* bhh1  = (const float*)d_in[11];
    float* out = (float*)d_out;                   // (B,T,V) f32

    // ---- workspace carve-out ----
    size_t off = 0;
    auto carve = [&](size_t bytes) -> void* {
        void* p = (char*)d_ws + off;
        off += (bytes + 255) & ~(size_t)255;
        return p;
    };
    unsigned short* Wih0b  = (unsigned short*)carve((size_t)CG * CH * 2);
    unsigned short* Whh0b  = (unsigned short*)carve((size_t)CG * CH * 2);
    unsigned short* Wih1b  = (unsigned short*)carve((size_t)CG * CH * 2);
    unsigned short* Whh1b  = (unsigned short*)carve((size_t)CG * CH * 2);
    unsigned short* Wprojb = (unsigned short*)carve((size_t)CV * CH * 2);
    unsigned short* Xbf    = (unsigned short*)carve((size_t)CM * CH * 2);
    float*          PRE    = (float*)         carve((size_t)CM * CG * 4);
    unsigned short* H0     = (unsigned short*)carve((size_t)CM * CH * 2);
    unsigned short* H1     = (unsigned short*)carve((size_t)CM * CH * 2);
    unsigned short* hstate = (unsigned short*)carve((size_t)16 * CH * 2);
    unsigned*       bar    = (unsigned*)      carve(256);
    (void)ws_size; (void)in_sizes; (void)n_in; (void)out_size;

    // Allow >64KB dynamic LDS for the recurrence kernel (host-side, capture-safe).
    (void)hipFuncSetAttribute(reinterpret_cast<const void*>(k_lstm),
                              hipFuncAttributeMaxDynamicSharedMemorySize,
                              LSTM_SMEM_BYTES);

    // ---- 1) convert weights to bf16 ----
    {
        dim3 blk(256), grd(2048);
        k_cvt_bf16<<<grd, blk, 0, stream>>>(Wih0,  Wih0b,  CG * CH);
        k_cvt_bf16<<<grd, blk, 0, stream>>>(Whh0,  Whh0b,  CG * CH);
        k_cvt_bf16<<<grd, blk, 0, stream>>>(Wih1,  Wih1b,  CG * CH);
        k_cvt_bf16<<<grd, blk, 0, stream>>>(Whh1,  Whh1b,  CG * CH);
        k_cvt_bf16<<<grd, blk, 0, stream>>>(Wproj, Wprojb, CV * CH);
    }

    // ---- 2) embedding gather -> bf16 ----
    k_gather<<<dim3(CM), dim3(256), 0, stream>>>(x, embed, Xbf);

    // ---- 3) layer 0: PRE = X @ Wih0^T + bih0 + bhh0 ----
    k_gemm<<<dim3(CG / 128, CM / 128), dim3(256), 0, stream>>>(Xbf, Wih0b, bih0, bhh0, PRE, CG);

    // ---- 4) layer 0 recurrence ----
    hipMemsetAsync(hstate, 0, (size_t)16 * CH * 2, stream);
    hipMemsetAsync(bar, 0, 256, stream);
    k_lstm<<<dim3(NWG_LSTM), dim3(256), LSTM_SMEM_BYTES, stream>>>(PRE, Whh0b, hstate, H0, bar);

    // ---- 5) layer 1: PRE = H0 @ Wih1^T + bih1 + bhh1 ----
    k_gemm<<<dim3(CG / 128, CM / 128), dim3(256), 0, stream>>>(H0, Wih1b, bih1, bhh1, PRE, CG);

    // ---- 6) layer 1 recurrence ----
    hipMemsetAsync(hstate, 0, (size_t)16 * CH * 2, stream);
    hipMemsetAsync(bar, 0, 256, stream);
    k_lstm<<<dim3(NWG_LSTM), dim3(256), LSTM_SMEM_BYTES, stream>>>(PRE, Whh1b, hstate, H1, bar);

    // ---- 7) logits = H1 @ Wproj^T + bproj ----
    k_gemm<<<dim3(CV / 128, CM / 128), dim3(256), 0, stream>>>(H1, Wprojb, bproj, nullptr, out, CV);
}
